// XMIX_4990751998354
// MI455X (gfx1250) — matrix operand order repaired and checked by reading, not compiled, not run on hardware
//
#include <hip/hip_runtime.h>
#include <math.h>

// ---------------------------------------------------------------------------
// QMIX-style mixer, fused for MI455X (gfx1250, wave32, WMMA).
//   B = 128*256 = 32768 rows, N_AGENTS=8, FEAT=64, MIX_EMB=32, EMB=32,
//   HYP=256, STATE_DIM=512.
// Big GEMMs run on v_wmma_f32_16x16x32_f16 (f16 operands, fp32 accumulate).
// Weights are pre-packed into WMMA B-fragment register order so each
// fragment is a single contiguous 32-byte load (2x global_load_b128).
// ---------------------------------------------------------------------------

typedef __attribute__((ext_vector_type(16))) _Float16 v16h;
typedef __attribute__((ext_vector_type(8)))  float    v8f;

#define BTOT   32768          // BS*T
#define ROWS   16             // rows per workgroup (one wave)
#define YSTR   520            // LDS stride for y tile (512 + pad)
#define HSTR   264            // LDS stride for 256-wide tiles (256 + pad)

// ---- pack f32 weight (K x N) into f16 WMMA B-fragment order ---------------
// B fragment mapping (32x16 K-x-N tile, wave32):
//   lane l: n = nt*16 + (l & 15), k = kt*32 + ((l>>4)<<3) + ((j>>3)<<4) + (j&7)   (j = 0..15)
// packed[((nt*KT + kt)*32 + l)*16 + j] = W[k][n]
__global__ void pack_b_f16(const float* __restrict__ src,
                           _Float16* __restrict__ dst, int K, int N) {
  int o = blockIdx.x * blockDim.x + threadIdx.x;
  if (o >= K * N) return;
  int KT = K >> 5;
  int j  = o & 15;
  int l  = (o >> 4) & 31;
  int t  = o >> 9;
  int kt = t % KT;
  int nt = t / KT;
  int k  = kt * 32 + ((l >> 4) << 3) + ((j >> 3) << 4) + (j & 7);
  int n  = nt * 16 + (l & 15);
  dst[o] = (_Float16)src[k * N + n];
}

// ---- WMMA fragment loaders -----------------------------------------------
// A (16x32 f16, M x K) from LDS: lanes 0-15 hold M=0..15, lanes 16-31 same M
// with upper K half-block; VGPR j maps K = ((j>>3)<<4) + (j&7) + 8*(lane>=16).
// The two 8-half runs are contiguous -> compiler emits 2x ds_load_b128.
__device__ __forceinline__ v16h ldsA(const _Float16* base, int stride, int koff) {
  int lane = threadIdx.x & 31;
  int m    = lane & 15;
  int kh   = (lane >> 4) << 3;
  const _Float16* p = base + m * stride + koff + kh;
  v16h a;
#pragma unroll
  for (int j = 0; j < 16; ++j) a[j] = p[((j >> 3) << 4) + (j & 7)];
  return a;
}

// B fragment from pre-packed weights: one contiguous 32-byte load per lane.
__device__ __forceinline__ v16h ldpB(const _Float16* __restrict__ P, int KT,
                                     int kt, int nt) {
  int lane = threadIdx.x & 31;
  return *(const v16h*)(P + (((size_t)(nt * KT + kt) * 32 + lane) << 4));
}

#define WMMA_F16(acc, a, b)                                                  \
  __builtin_amdgcn_wmma_f32_16x16x32_f16(false, (a), false, (b), (short)0,   \
                                         (acc), false, false)

// GEMM inner loop over K with two independent accumulator chains so
// back-to-back v_wmma ops have no RAW dependency on the accumulator.
#define GEMM_K(accv, Abase, Astride, Wpack, KT, nt)                          \
  {                                                                          \
    v8f acc0 = {}, acc1 = {};                                                \
    _Pragma("unroll 1")                                                      \
    for (int kt = 0; kt < (KT); kt += 2) {                                   \
      v16h a0 = ldsA((Abase), (Astride), kt * 32);                           \
      v16h b0 = ldpB((Wpack), (KT), kt, (nt));                               \
      v16h a1 = ldsA((Abase), (Astride), kt * 32 + 32);                      \
      v16h b1 = ldpB((Wpack), (KT), kt + 1, (nt));                           \
      acc0 = WMMA_F16(acc0, a0, b0);                                         \
      acc1 = WMMA_F16(acc1, a1, b1);                                         \
    }                                                                        \
    _Pragma("unroll")                                                        \
    for (int i = 0; i < 8; ++i) accv[i] = acc0[i] + acc1[i];                 \
  }

// ---------------------------------------------------------------------------
__global__ void __launch_bounds__(32)
mixer_kernel(const float* __restrict__ agent_qs,   // (B,8)
             const float* __restrict__ states,     // (B,512)
             const float* __restrict__ W_tok,      // (64,32)
             const float* __restrict__ b_tok,      // (32)
             const float* __restrict__ W_yfc,      // (32,64)
             const float* __restrict__ b_yfc,      // (64)
             const float* __restrict__ b1a,        // (256)
             const float* __restrict__ b1b,        // (256)
             const float* __restrict__ bfa,        // (256)
             const float* __restrict__ bfb,        // (32)
             const float* __restrict__ bb1,        // (32)
             const float* __restrict__ bv1,        // (32)
             const float* __restrict__ Wv2,        // (32,1)
             const float* __restrict__ bv2,        // (1)
             const _Float16* __restrict__ W1a_p,   // packed (512,256)
             const _Float16* __restrict__ W1b_p,   // packed (256,256)
             const _Float16* __restrict__ Wfa_p,   // packed (512,256)
             const _Float16* __restrict__ Wfb_p,   // packed (256,32)
             const _Float16* __restrict__ Wb1_p,   // packed (512,32)
             const _Float16* __restrict__ Wv1_p,   // packed (512,32)
             float* __restrict__ out_qtot,         // (B)
             float* __restrict__ out_grad)         // (B,8)
{
  __shared__ float    sWtok[64 * 32];
  __shared__ float    sWyfc[32 * 64];
  __shared__ _Float16 yt[ROWS][YSTR];      // y tile, f16
  __shared__ float    es[8][33];           // per-row e scratch
  __shared__ _Float16 Hbuf[ROWS][HSTR];    // H1 then Hf, f16
  __shared__ _Float16 w1buf[ROWS][HSTR];   // |H1@W1b|, f16
  __shared__ float    wfbuf[ROWS][32];     // w_final
  __shared__ float    b1buf[ROWS][32];     // b1
  __shared__ float    hvbuf[ROWS][32];     // relu(y@Wv1)
  __shared__ float    vbuf[ROWS];          // v
  __shared__ float    tbuf[32];            // delu*w_final scratch

  const int lane = threadIdx.x;            // 0..31, one wave per block
  const int row0 = blockIdx.x * ROWS;

  // ---------------- Phase A: token MLP -> y tile in LDS (f16) -------------
  for (int i = lane; i < 64 * 32; i += 32) sWtok[i] = W_tok[i];
  for (int i = lane; i < 32 * 64; i += 32) sWyfc[i] = W_yfc[i];
  __syncthreads();

  for (int m = 0; m < ROWS; ++m) {
    const float* srow = states + (size_t)(row0 + m) * 512;
    float e[8];
#pragma unroll
    for (int a = 0; a < 8; ++a) e[a] = 0.f;
    for (int f = 0; f < 64; ++f) {
      float w = sWtok[f * 32 + lane];
#pragma unroll
      for (int a = 0; a < 8; ++a) e[a] += srow[a * 64 + f] * w;
    }
    float bt = b_tok[lane];
#pragma unroll
    for (int a = 0; a < 8; ++a) {
      float v = e[a] + bt;
      es[a][lane] = v > 0.f ? v : 0.f;
    }
    __syncthreads();
    float by0 = b_yfc[lane], by1 = b_yfc[32 + lane];
#pragma unroll
    for (int a = 0; a < 8; ++a) {
      float y0 = by0, y1 = by1;
      for (int j = 0; j < 32; ++j) {
        float ev = es[a][j];
        y0 += ev * sWyfc[j * 64 + lane];
        y1 += ev * sWyfc[j * 64 + 32 + lane];
      }
      yt[m][a * 64 + lane]      = (_Float16)(y0 > 0.f ? y0 : 0.f);
      yt[m][a * 64 + 32 + lane] = (_Float16)(y1 > 0.f ? y1 : 0.f);
    }
    __syncthreads();
  }

  const int cn   = lane & 15;          // C-fragment column
  const int mrow = (lane >> 4) << 3;   // C-fragment row base (0 or 8)

  // ---------------- H1 = relu(y @ W1a + b1a)  (16x256, K=512) -------------
#pragma unroll 1
  for (int nt = 0; nt < 16; ++nt) {
    v8f acc;
    GEMM_K(acc, &yt[0][0], YSTR, W1a_p, 16, nt);
    float bias = b1a[nt * 16 + cn];
#pragma unroll
    for (int i = 0; i < 8; ++i) {
      float v = acc[i] + bias;
      Hbuf[mrow + i][nt * 16 + cn] = (_Float16)(v > 0.f ? v : 0.f);
    }
  }
  __syncthreads();

  // ---------------- w1 = |H1 @ W1b + b1b|  (16x256, K=256) ----------------
#pragma unroll 1
  for (int nt = 0; nt < 16; ++nt) {
    v8f acc;
    GEMM_K(acc, &Hbuf[0][0], HSTR, W1b_p, 8, nt);
    float bias = b1b[nt * 16 + cn];
#pragma unroll
    for (int i = 0; i < 8; ++i)
      w1buf[mrow + i][nt * 16 + cn] = (_Float16)fabsf(acc[i] + bias);
  }
  __syncthreads();

  // ---------------- Hf = relu(y @ Wfa + bfa)  (overwrite Hbuf) ------------
#pragma unroll 1
  for (int nt = 0; nt < 16; ++nt) {
    v8f acc;
    GEMM_K(acc, &yt[0][0], YSTR, Wfa_p, 16, nt);
    float bias = bfa[nt * 16 + cn];
#pragma unroll
    for (int i = 0; i < 8; ++i) {
      float v = acc[i] + bias;
      Hbuf[mrow + i][nt * 16 + cn] = (_Float16)(v > 0.f ? v : 0.f);
    }
  }
  __syncthreads();

  // ---------------- w_final = |Hf @ Wfb + bfb|  (16x32, K=256) ------------
#pragma unroll 1
  for (int nt = 0; nt < 2; ++nt) {
    v8f acc;
    GEMM_K(acc, &Hbuf[0][0], HSTR, Wfb_p, 8, nt);
    float bias = bfb[nt * 16 + cn];
#pragma unroll
    for (int i = 0; i < 8; ++i)
      wfbuf[mrow + i][nt * 16 + cn] = fabsf(acc[i] + bias);
  }

  // ---------------- b1 = y @ Wb1 + bb1  (16x32, K=512) --------------------
#pragma unroll 1
  for (int nt = 0; nt < 2; ++nt) {
    v8f acc;
    GEMM_K(acc, &yt[0][0], YSTR, Wb1_p, 16, nt);
    float bias = bb1[nt * 16 + cn];
#pragma unroll
    for (int i = 0; i < 8; ++i)
      b1buf[mrow + i][nt * 16 + cn] = acc[i] + bias;
  }

  // ---------------- Hv = relu(y @ Wv1 + bv1)  (16x32, K=512) --------------
#pragma unroll 1
  for (int nt = 0; nt < 2; ++nt) {
    v8f acc;
    GEMM_K(acc, &yt[0][0], YSTR, Wv1_p, 16, nt);
    float bias = bv1[nt * 16 + cn];
#pragma unroll
    for (int i = 0; i < 8; ++i) {
      float v = acc[i] + bias;
      hvbuf[mrow + i][nt * 16 + cn] = v > 0.f ? v : 0.f;
    }
  }
  __syncthreads();

  // ---------------- v = Hv @ Wv2 + bv2  (16x1) ----------------------------
  if (lane < ROWS) {
    float acc = bv2[0];
    for (int e = 0; e < 32; ++e) acc += hvbuf[lane][e] * Wv2[e];
    vbuf[lane] = acc;
  }
  __syncthreads();

  // ---------------- Epilogue: hidden/elu, q_tot, delu, grad ---------------
  for (int m = 0; m < ROWS; ++m) {
    const float* qrow = agent_qs + (size_t)(row0 + m) * 8;
    float q[8];
#pragma unroll
    for (int a = 0; a < 8; ++a) q[a] = qrow[a];

    int e = lane;
    float h = b1buf[m][e];
#pragma unroll
    for (int a = 0; a < 8; ++a) h += q[a] * (float)w1buf[m][a * 32 + e];
    float hidden = h > 0.f ? h : (expf(h) - 1.f);
    float wf = wfbuf[m][e];

    float part = hidden * wf;                 // q_tot wave32 reduction
#pragma unroll
    for (int off = 16; off > 0; off >>= 1) part += __shfl_xor(part, off, 32);
    if (lane == 0) out_qtot[row0 + m] = part + vbuf[m];

    float delu = hidden < 0.f ? expf(hidden) : 1.f;
    tbuf[e] = delu * wf;
    __syncthreads();

    if (lane < 8) {
      float g = 0.f;
      for (int ee = 0; ee < 32; ++ee)
        g += (float)w1buf[m][lane * 32 + ee] * tbuf[ee];
      out_grad[(size_t)(row0 + m) * 8 + lane] = g;
    }
    __syncthreads();
  }
}

// ---------------------------------------------------------------------------
extern "C" void kernel_launch(void* const* d_in, const int* in_sizes, int n_in,
                              void* d_out, int out_size, void* d_ws, size_t ws_size,
                              hipStream_t stream) {
  (void)in_sizes; (void)n_in; (void)out_size; (void)ws_size;

  const float* agent_qs = (const float*)d_in[0];
  // d_in[1] = hist (unused by reference)
  const float* states   = (const float*)d_in[2];
  // d_in[3] = obs (unused by reference)
  const float* W_tok = (const float*)d_in[4];
  const float* b_tok = (const float*)d_in[5];
  const float* W_yfc = (const float*)d_in[6];
  const float* b_yfc = (const float*)d_in[7];
  const float* W1a   = (const float*)d_in[8];
  const float* b1a   = (const float*)d_in[9];
  const float* W1b   = (const float*)d_in[10];
  const float* b1b   = (const float*)d_in[11];
  const float* Wfa   = (const float*)d_in[12];
  const float* bfa   = (const float*)d_in[13];
  const float* Wfb   = (const float*)d_in[14];
  const float* bfb   = (const float*)d_in[15];
  const float* Wb1   = (const float*)d_in[16];
  const float* bb1   = (const float*)d_in[17];
  const float* Wv1   = (const float*)d_in[18];
  const float* bv1   = (const float*)d_in[19];
  const float* Wv2   = (const float*)d_in[20];
  const float* bv2   = (const float*)d_in[21];

  // f16 packed-weight copies in workspace
  _Float16* ws = (_Float16*)d_ws;
  _Float16* W1a_p = ws;                 // 512*256
  _Float16* W1b_p = ws + 131072;        // 256*256
  _Float16* Wfa_p = ws + 196608;        // 512*256
  _Float16* Wfb_p = ws + 327680;        // 256*32
  _Float16* Wb1_p = ws + 335872;        // 512*32
  _Float16* Wv1_p = ws + 352256;        // 512*32

  pack_b_f16<<<(131072 + 255) / 256, 256, 0, stream>>>(W1a, W1a_p, 512, 256);
  pack_b_f16<<<(65536  + 255) / 256, 256, 0, stream>>>(W1b, W1b_p, 256, 256);
  pack_b_f16<<<(131072 + 255) / 256, 256, 0, stream>>>(Wfa, Wfa_p, 512, 256);
  pack_b_f16<<<(8192   + 255) / 256, 256, 0, stream>>>(Wfb, Wfb_p, 256, 32);
  pack_b_f16<<<(16384  + 255) / 256, 256, 0, stream>>>(Wb1, Wb1_p, 512, 32);
  pack_b_f16<<<(16384  + 255) / 256, 256, 0, stream>>>(Wv1, Wv1_p, 512, 32);

  float* out_qtot = (float*)d_out;          // B values
  float* out_grad = out_qtot + BTOT;        // B*8 values

  mixer_kernel<<<BTOT / ROWS, 32, 0, stream>>>(
      agent_qs, states, W_tok, b_tok, W_yfc, b_yfc,
      b1a, b1b, bfa, bfb, bb1, bv1, Wv2, bv2,
      W1a_p, W1b_p, Wfa_p, Wfb_p, Wb1_p, Wv1_p,
      out_qtot, out_grad);
}
